// MultiHeadAttention_85031762526692
// MI455X (gfx1250) — matrix operand order repaired and checked by reading, not compiled, not run on hardware
//
#include <hip/hip_runtime.h>
#include <hip/hip_bf16.h>
#include <math.h>

// ---------------------------------------------------------------------------
// Types (trivially-constructible vectors so they can live in unions)
// ---------------------------------------------------------------------------
typedef __attribute__((ext_vector_type(16))) __bf16 v16bf;
typedef __attribute__((ext_vector_type(8)))  __bf16 bf16x8;
typedef __attribute__((ext_vector_type(8)))  float  v8f;
typedef __attribute__((ext_vector_type(8)))  float  f32x8;
typedef __attribute__((ext_vector_type(4)))  unsigned int u32x4;
typedef __attribute__((ext_vector_type(2)))  unsigned int u32x2;
typedef __attribute__((ext_vector_type(4)))  int    i32x4;

#define H_HEADS 16
#define A_DIM   64
#define D_MODEL 1024
#define B_BATCH 2
#define S_SEQ   2048
#define NTOK    (B_BATCH * S_SEQ)     // 4096
#define N_QKV   (3 * H_HEADS * A_DIM) // 3072
#define NEGV    (-1000000000.0f)

// ---------------------------------------------------------------------------
// Native bf16 conversion helpers (backend emits v_cvt_pk_bf16_f32)
// ---------------------------------------------------------------------------
__device__ __forceinline__ unsigned short bfbits(float x) {
    union { __bf16 h; unsigned short u; } c;
    c.h = (__bf16)x;
    return c.u;
}
__device__ __forceinline__ unsigned int pack_bf16x2(float a, float b) {
    union { __bf16 h[2]; unsigned int u; } c;
    c.h[0] = (__bf16)a; c.h[1] = (__bf16)b;
    return c.u;
}
__device__ __forceinline__ u32x4 pack_bf16x8(f32x8 f) {
    union { bf16x8 b; u32x4 q; } c;
    c.b = __builtin_convertvector(f, bf16x8);
    return c.q;
}
__device__ __forceinline__ v8f v8f_zero() {
    v8f z = {0.f, 0.f, 0.f, 0.f, 0.f, 0.f, 0.f, 0.f};
    return z;
}

// ---------------------------------------------------------------------------
// CDNA5 async global->LDS staging (ASYNCcnt-tracked).
// Signature learned from hipcc diagnostic: (int4 as1* src, int4 as3* dst, Ii, Ii)
// ---------------------------------------------------------------------------
#if defined(__has_builtin)
#if __has_builtin(__builtin_amdgcn_global_load_async_to_lds_b128) && \
    __has_builtin(__builtin_amdgcn_s_wait_asynccnt)
#define HAVE_ASYNC_LDS 1
#endif
#endif
#ifndef HAVE_ASYNC_LDS
#define HAVE_ASYNC_LDS 0
#endif

#if HAVE_ASYNC_LDS
typedef i32x4 __attribute__((address_space(1)))* as1_i32x4;
typedef i32x4 __attribute__((address_space(3)))* as3_i32x4;
__device__ __forceinline__ void async_copy_b128(const void* g, void* l) {
    __builtin_amdgcn_global_load_async_to_lds_b128(
        (as1_i32x4)(unsigned long long)(size_t)g,
        (as3_i32x4)(unsigned int)(size_t)l, 0, 0);
}
__device__ __forceinline__ void async_wait0() {
    __builtin_amdgcn_s_wait_asynccnt(0);
}
#endif

// WMMA operand: 16 bf16 per lane == two 16-byte chunks
union ABOp {
    u32x4 q[2];
    v16bf v;
};

__device__ __forceinline__ v8f wmma_bf16(const ABOp& a, const ABOp& b, v8f c) {
    // D(16x16 f32) = A(16x32 bf16) * B(32x16 bf16) + C
    return __builtin_amdgcn_wmma_f32_16x16x32_bf16(
        /*neg_a=*/false, a.v, /*neg_b=*/false, b.v,
        /*c_mod=*/(short)0, c, /*reuse_a=*/false, /*reuse_b=*/false);
}

// ---------------------------------------------------------------------------
// Kernel 1: qkv = X(4096x1024,f32) @ Wqkv(1024x3072,f32), scattered to
//           Q/K/V (B,H,S,A) bf16. f32->bf16 conversion fused into staging.
// Block: 256 threads (8 waves). Tile: 128(M) x 128(N) x 32(K).
// ---------------------------------------------------------------------------
__global__ __launch_bounds__(256)
void qkv_gemm_kernel(const float* __restrict__ X, const float* __restrict__ W,
                     unsigned short* __restrict__ Qb,
                     unsigned short* __restrict__ Kb,
                     unsigned short* __restrict__ Vb) {
    // padded rows: 32 payload + 8 pad bf16 => 80B stride (16B aligned, bank-spread)
    __shared__ __align__(32) unsigned short At[128 * 40]; // A tile, row-major (M x K)
    __shared__ __align__(32) unsigned short Bt[128 * 40]; // B tile, transposed (N x K)

    const int tid  = threadIdx.x;
    const int lane = tid & 31;
    const int w    = tid >> 5;
    const int m0   = blockIdx.y * 128;
    const int n0   = blockIdx.x * 128;

    v8f acc[8];
#pragma unroll
    for (int i = 0; i < 8; ++i) acc[i] = v8f_zero();

    for (int k0 = 0; k0 < D_MODEL; k0 += 32) {
        { // stage A (vector convert f32->bf16): thread = 16 contiguous elems
            const int row = tid >> 1, half = tid & 1;
            const f32x8* src = (const f32x8*)(X + (size_t)(m0 + row) * D_MODEL + k0 + half * 16);
            u32x4* dst = (u32x4*)(At + row * 40 + half * 16);
            dst[0] = pack_bf16x8(src[0]);
            dst[1] = pack_bf16x8(src[1]);
            if (k0 + 32 < D_MODEL)
                __builtin_prefetch((const float*)src + 32, 0, 0);
        }
        { // stage B transposed: thread handles k-pair (2 rows) x 8 cols,
          // packed b32 stores along the contiguous k direction of Bt
            const int kp = tid & 15;   // k pair: k = 2kp, 2kp+1
            const int ng = tid >> 4;   // n group: n = ng*8 .. ng*8+7
            const float* s0 = W + (size_t)(k0 + 2 * kp) * N_QKV + n0 + ng * 8;
            f32x8 ra = *(const f32x8*)s0;
            f32x8 rb = *(const f32x8*)(s0 + N_QKV);
#pragma unroll
            for (int i = 0; i < 8; ++i)
                *(unsigned int*)(Bt + (ng * 8 + i) * 40 + 2 * kp) = pack_bf16x2(ra[i], rb[i]);
            if (k0 + 32 < D_MODEL)
                __builtin_prefetch(s0 + 32 * N_QKV, 0, 0);
        }
        __syncthreads();

        // A operand (ISA A layout: row = lane&15, K chunks kh*8+{0..7,16..23})
        ABOp aop;
        {
            const int r = lane & 15, kh = lane >> 4;
            const unsigned short* base = At + (w * 16 + r) * 40 + kh * 8;
            aop.q[0] = *(const u32x4*)(base);
            aop.q[1] = *(const u32x4*)(base + 16);
        }
        // B operands preloaded in groups of 4 so ds_loads pipeline ahead of WMMAs
#pragma unroll
        for (int g = 0; g < 2; ++g) {
            ABOp bop[4];
#pragma unroll
            for (int n = 0; n < 4; ++n) {
                const unsigned short* base =
                    Bt + ((g * 4 + n) * 16 + (lane & 15)) * 40 + (lane >> 4) * 8;
                bop[n].q[0] = *(const u32x4*)(base);
                bop[n].q[1] = *(const u32x4*)(base + 16);
            }
#pragma unroll
            for (int n = 0; n < 4; ++n)
                acc[g * 4 + n] = wmma_bf16(aop, bop[n], acc[g * 4 + n]);
        }
        __syncthreads();
    }

    // Epilogue: scatter into Q/K/V (B,H,S,A) bf16
#pragma unroll
    for (int n = 0; n < 8; ++n) {
        const int col = n0 + n * 16 + (lane & 15);   // 0..3071
        const int sec = col >> 10;                   // 0=Q 1=K 2=V
        const int rem = col & 1023;
        const int h = rem >> 6, a = rem & 63;
        unsigned short* dstb = (sec == 0) ? Qb : ((sec == 1) ? Kb : Vb);
#pragma unroll
        for (int v = 0; v < 8; ++v) {
            const int row = m0 + w * 16 + v + ((lane >> 4) << 3); // token id
            const int b = row >> 11, s = row & 2047;
            const size_t dst = ((((size_t)b * H_HEADS + h) * S_SEQ) + s) * A_DIM + a;
            dstb[dst] = bfbits(acc[n][v]);
        }
    }
}

// ---------------------------------------------------------------------------
// Kernel 2: flash attention. Block = 8 waves, one (b,h), 128 q rows.
// KV step = 32. Scores: 4 WMMAs per step; PV: 4 WMMAs per step.
// ---------------------------------------------------------------------------
__global__ __launch_bounds__(256)
void attn_kernel(const unsigned short* __restrict__ Qb,
                 const unsigned short* __restrict__ Kb,
                 const unsigned short* __restrict__ Vb,
                 const int* __restrict__ mask,
                 unsigned short* __restrict__ Ctx) {
    __shared__ __align__(32) unsigned short Kt[32 * 72];      // K tile (kv x a), 144B stride
    __shared__ __align__(32) unsigned short Vt[64 * 40];      // V^T tile (a x kv), 80B stride
    __shared__ __align__(32) unsigned short Pb[8 * 16 * 40];  // per-wave P buffer

    const int tid = threadIdx.x, lane = tid & 31, w = tid >> 5;
    const int bh = blockIdx.y;          // b*H + h
    const int b  = bh >> 4;
    const int h  = bh & 15;
    const int q0 = blockIdx.x * 128;
    const size_t headbase = (size_t)bh * S_SEQ * A_DIM;

    // Q operands for this wave's 16 rows (contraction a split into 2x32)
    ABOp qop[2];
    {
        const int r = lane & 15, kh = lane >> 4;
        const unsigned short* qrow = Qb + headbase + (size_t)(q0 + w * 16 + r) * A_DIM;
#pragma unroll
        for (int st = 0; st < 2; ++st) {
            qop[st].q[0] = *(const u32x4*)(qrow + st * 32 + kh * 8);
            qop[st].q[1] = *(const u32x4*)(qrow + st * 32 + kh * 8 + 16);
        }
    }

    v8f o[4];
#pragma unroll
    for (int t = 0; t < 4; ++t) o[t] = v8f_zero();
    float mrow[8], lrow[8];
#pragma unroll
    for (int v = 0; v < 8; ++v) { mrow[v] = -INFINITY; lrow[v] = 0.f; }

    for (int j = 0; j < S_SEQ; j += 32) {
        { // stage K tile row-major (async b128 path)
            const int row = tid >> 3, seg = tid & 7;
            const unsigned short* ksrc = Kb + headbase + (size_t)(j + row) * A_DIM + seg * 8;
#if HAVE_ASYNC_LDS
            async_copy_b128(ksrc, Kt + row * 72 + seg * 8);
#else
            *(u32x4*)(Kt + row * 72 + seg * 8) = *(const u32x4*)ksrc;
#endif
            if (j + 32 < S_SEQ)
                __builtin_prefetch(ksrc + 32 * A_DIM, 0, 0);
        }
        { // stage V^T: thread = kv-pair x 4 a-values -> packed b32 LDS stores
            const int kvp = tid & 15;   // kv rows 2kvp, 2kvp+1
            const int ag  = tid >> 4;   // a values ag*4 .. +3
            const unsigned short* vsrc = Vb + headbase + (size_t)(j + 2 * kvp) * A_DIM + ag * 4;
            u32x2 va = *(const u32x2*)vsrc;
            u32x2 vb = *(const u32x2*)(vsrc + A_DIM);
            const unsigned short* pa = (const unsigned short*)&va;
            const unsigned short* pb = (const unsigned short*)&vb;
#pragma unroll
            for (int i = 0; i < 4; ++i)
                *(unsigned int*)(Vt + (ag * 4 + i) * 40 + 2 * kvp) =
                    (unsigned int)pa[i] | ((unsigned int)pb[i] << 16);
            if (j + 32 < S_SEQ)
                __builtin_prefetch(vsrc + 32 * A_DIM, 0, 0);
        }
#if HAVE_ASYNC_LDS
        async_wait0();
#endif
        __syncthreads();

        // scores: preload all 4 K-operands, then 4 WMMAs
        ABOp kop[4]; // [st*2 + tile]
#pragma unroll
        for (int st = 0; st < 2; ++st) {
            const unsigned short* kb0 = Kt + (lane & 15) * 72 + st * 32 + (lane >> 4) * 8;
            kop[st * 2].q[0] = *(const u32x4*)kb0;
            kop[st * 2].q[1] = *(const u32x4*)(kb0 + 16);
            const unsigned short* kb1 = Kt + (16 + (lane & 15)) * 72 + st * 32 + (lane >> 4) * 8;
            kop[st * 2 + 1].q[0] = *(const u32x4*)kb1;
            kop[st * 2 + 1].q[1] = *(const u32x4*)(kb1 + 16);
        }
        v8f s0 = v8f_zero(), s1 = v8f_zero();
        s0 = wmma_bf16(qop[0], kop[0], s0);
        s1 = wmma_bf16(qop[0], kop[1], s1);
        s0 = wmma_bf16(qop[1], kop[2], s0);
        s1 = wmma_bf16(qop[1], kop[3], s1);

        // scale + mask (mask depends only on kv column = lane-uniform over v)
        const int mv0 = mask[b * S_SEQ + j + (lane & 15)];
        const int mv1 = mask[b * S_SEQ + j + 16 + (lane & 15)];
        float s0v[8], s1v[8], lm[8];
#pragma unroll
        for (int v = 0; v < 8; ++v) {
            s0v[v] = mv0 ? s0[v] * 0.125f : NEGV;  // 1/sqrt(64)
            s1v[v] = mv1 ? s1[v] * 0.125f : NEGV;
            lm[v] = fmaxf(s0v[v], s1v[v]);
        }
        // row max across the 16 lanes holding this row (xor<16 stays in half)
#pragma unroll
        for (int off = 1; off < 16; off <<= 1) {
#pragma unroll
            for (int v = 0; v < 8; ++v) lm[v] = fmaxf(lm[v], __shfl_xor(lm[v], off, 32));
        }
        float alpha[8], p0[8], p1[8], rs[8];
#pragma unroll
        for (int v = 0; v < 8; ++v) {
            const float mn = fmaxf(mrow[v], lm[v]);
            alpha[v] = __expf(mrow[v] - mn);
            mrow[v] = mn;
            p0[v] = __expf(s0v[v] - mn);
            p1[v] = __expf(s1v[v] - mn);
            rs[v] = p0[v] + p1[v];
        }
#pragma unroll
        for (int off = 1; off < 16; off <<= 1) {
#pragma unroll
            for (int v = 0; v < 8; ++v) rs[v] += __shfl_xor(rs[v], off, 32);
        }
#pragma unroll
        for (int v = 0; v < 8; ++v) lrow[v] = lrow[v] * alpha[v] + rs[v];
#pragma unroll
        for (int t = 0; t < 4; ++t)
#pragma unroll
            for (int v = 0; v < 8; ++v) o[t][v] *= alpha[v];

        // P (C layout) -> LDS -> A-operand layout
        unsigned short* pw = Pb + w * 16 * 40;
#pragma unroll
        for (int v = 0; v < 8; ++v) {
            const int r = v + ((lane >> 4) << 3);
            pw[r * 40 + (lane & 15)]      = bfbits(p0[v]);
            pw[r * 40 + 16 + (lane & 15)] = bfbits(p1[v]);
        }
        __syncthreads();   // make P visible cross-lane (also closes Kt reads)

        ABOp pop;
        {
            const int r = lane & 15, kh = lane >> 4;
            const unsigned short* base = pw + r * 40 + kh * 8;
            pop.q[0] = *(const u32x4*)base;
            pop.q[1] = *(const u32x4*)(base + 16);
        }
        // PV: preload all 4 V operands, then 4 WMMAs
        ABOp vop[4];
#pragma unroll
        for (int t = 0; t < 4; ++t) {
            const unsigned short* base = Vt + (t * 16 + (lane & 15)) * 40 + (lane >> 4) * 8;
            vop[t].q[0] = *(const u32x4*)base;
            vop[t].q[1] = *(const u32x4*)(base + 16);
        }
#pragma unroll
        for (int t = 0; t < 4; ++t)
            o[t] = wmma_bf16(pop, vop[t], o[t]);
        __syncthreads();   // Vt/Pb reads done before next restage
    }

    // finalize and store ctx (B,S,H*A) bf16
#pragma unroll
    for (int v = 0; v < 8; ++v) lrow[v] = 1.0f / lrow[v];
#pragma unroll
    for (int t = 0; t < 4; ++t) {
#pragma unroll
        for (int v = 0; v < 8; ++v) {
            const int q = q0 + w * 16 + v + ((lane >> 4) << 3);
            const size_t dst = ((size_t)b * S_SEQ + q) * D_MODEL + h * A_DIM + t * 16 + (lane & 15);
            Ctx[dst] = bfbits(o[t][v] * lrow[v]);
        }
    }
}

// ---------------------------------------------------------------------------
// Kernel 3: out = ctx(4096x1024,bf16) @ Wout(1024x1024,f32->bf16) + bout, f32
// ---------------------------------------------------------------------------
__global__ __launch_bounds__(256)
void out_gemm_kernel(const unsigned short* __restrict__ Ctx,
                     const float* __restrict__ W,
                     const float* __restrict__ bias,
                     float* __restrict__ out) {
    __shared__ __align__(32) unsigned short At[128 * 40];
    __shared__ __align__(32) unsigned short Bt[128 * 40];

    const int tid  = threadIdx.x;
    const int lane = tid & 31;
    const int w    = tid >> 5;
    const int m0   = blockIdx.y * 128;
    const int n0   = blockIdx.x * 128;

    v8f acc[8];
#pragma unroll
    for (int i = 0; i < 8; ++i) acc[i] = v8f_zero();

    for (int k0 = 0; k0 < D_MODEL; k0 += 32) {
        { // stage A (already bf16): async b128 copies
            const int row = tid >> 1, half = tid & 1;
            const unsigned short* src = Ctx + (size_t)(m0 + row) * D_MODEL + k0 + half * 16;
            unsigned short* dst = At + row * 40 + half * 16;
#if HAVE_ASYNC_LDS
            async_copy_b128(src, dst);
            async_copy_b128(src + 8, dst + 8);
#else
            *(u32x4*)(dst)     = *(const u32x4*)(src);
            *(u32x4*)(dst + 8) = *(const u32x4*)(src + 8);
#endif
            if (k0 + 32 < D_MODEL)
                __builtin_prefetch(src + 32, 0, 0);
        }
        { // stage B transposed, k-pair packed b32 stores
            const int kp = tid & 15;
            const int ng = tid >> 4;
            const float* s0 = W + (size_t)(k0 + 2 * kp) * D_MODEL + n0 + ng * 8;
            f32x8 ra = *(const f32x8*)s0;
            f32x8 rb = *(const f32x8*)(s0 + D_MODEL);
#pragma unroll
            for (int i = 0; i < 8; ++i)
                *(unsigned int*)(Bt + (ng * 8 + i) * 40 + 2 * kp) = pack_bf16x2(ra[i], rb[i]);
            if (k0 + 32 < D_MODEL)
                __builtin_prefetch(s0 + 32 * D_MODEL, 0, 0);
        }
#if HAVE_ASYNC_LDS
        async_wait0();
#endif
        __syncthreads();

        ABOp aop;
        {
            const int r = lane & 15, kh = lane >> 4;
            const unsigned short* base = At + (w * 16 + r) * 40 + kh * 8;
            aop.q[0] = *(const u32x4*)(base);
            aop.q[1] = *(const u32x4*)(base + 16);
        }
#pragma unroll
        for (int g = 0; g < 2; ++g) {
            ABOp bop[4];
#pragma unroll
            for (int n = 0; n < 4; ++n) {
                const unsigned short* base =
                    Bt + ((g * 4 + n) * 16 + (lane & 15)) * 40 + (lane >> 4) * 8;
                bop[n].q[0] = *(const u32x4*)(base);
                bop[n].q[1] = *(const u32x4*)(base + 16);
            }
#pragma unroll
            for (int n = 0; n < 4; ++n)
                acc[g * 4 + n] = wmma_bf16(aop, bop[n], acc[g * 4 + n]);
        }
        __syncthreads();
    }

#pragma unroll
    for (int n = 0; n < 8; ++n) {
        const int col = n0 + n * 16 + (lane & 15);
        const float bb = bias[col];
#pragma unroll
        for (int v = 0; v < 8; ++v) {
            const int row = m0 + w * 16 + v + ((lane >> 4) << 3);
            out[(size_t)row * D_MODEL + col] = acc[n][v] + bb;
        }
    }
}

// ---------------------------------------------------------------------------
// Launch
// ---------------------------------------------------------------------------
extern "C" void kernel_launch(void* const* d_in, const int* in_sizes, int n_in,
                              void* d_out, int out_size, void* d_ws, size_t ws_size,
                              hipStream_t stream) {
    const float* qs   = (const float*)d_in[0];  // (B,S,D)
    const int*   mask = (const int*)d_in[1];    // (B,S)
    const float* Wqkv = (const float*)d_in[2];  // (D, 3*H*A)
    const float* Wout = (const float*)d_in[3];  // (H*A, D)
    const float* bout = (const float*)d_in[4];  // (D,)
    float* out = (float*)d_out;                 // (B,S,D) f32

    // workspace: Q,K,V (B,H,S,A) bf16 + ctx (B,S,H*A) bf16  => 33.5 MB
    const size_t HEAD_ELEMS = (size_t)B_BATCH * H_HEADS * S_SEQ * A_DIM; // 4,194,304
    unsigned short* Qb  = (unsigned short*)d_ws;
    unsigned short* Kb  = Qb + HEAD_ELEMS;
    unsigned short* Vb  = Kb + HEAD_ELEMS;
    unsigned short* Ctx = Vb + HEAD_ELEMS;

    qkv_gemm_kernel<<<dim3(N_QKV / 128, NTOK / 128), 256, 0, stream>>>(qs, Wqkv, Qb, Kb, Vb);
    attn_kernel<<<dim3(S_SEQ / 128, B_BATCH * H_HEADS), 256, 0, stream>>>(Qb, Kb, Vb, mask, Ctx);
    out_gemm_kernel<<<dim3(D_MODEL / 128, NTOK / 128), 256, 0, stream>>>(Ctx, Wout, bout, out);
}
